// TimelineGNNLayer2_39410619908397
// MI455X (gfx1250) — matrix operand order repaired and checked by reading, not compiled, not run on hardware
//
#include <hip/hip_runtime.h>
#include <hip/hip_bf16.h>
#include <math.h>

typedef _Float16 h16;
typedef __attribute__((ext_vector_type(16))) _Float16 v16h;
typedef __attribute__((ext_vector_type(8)))  _Float16 v8h;
typedef __attribute__((ext_vector_type(8)))  float    v8f;

#define N_NODE   50000
#define N_EDGE   800000
#define BLK      128        // 4 wave32 waves per block

#define EDGE_BLOCKS 1250
#define EDGE_ITERS  10      // 1250 blocks * 10 iters * 64 edges = 800000

#define OUT_TILES   3125    // ceil(50000/16)
#define OUT_BLOCKS  196
#define OUT_BOUND   3136    // 196*4*4 : uniform 4 iterations per wave

static_assert(N_EDGE == EDGE_BLOCKS * EDGE_ITERS * 64, "edge tiling must cover exactly");

// ---------------- LDS layout (edge kernel), all offsets 16B aligned ----------
#define OFF_B1    0          // 128 f32
#define OFF_B2    512        // 128 f32
#define OFF_BQR   1024       // 64 f32
#define OFF_WA    1280       // 64 f32
#define OFF_W1T   1536       // 128x160 f16  (W1^T, [N][K])
#define OFF_W2T   42496      // 128x128 f16
#define OFF_WST   75264      // 64x128 f16
#define OFF_WRT   91648      // 64x128 f16
#define OFF_WQT   108032     // 64x128 f16
#define OFF_WAVE  124416
#define WAVE_STRIDE 17920
//   per-wave: comp[16][160]f16 @+0 (5120), hbuf[16][128]f16 @+5120 (4096),
//             hsbuf @+9216 (4096), hqrbuf @+13312 (4096),
//             ints[80] @+17408 (320), alpha[16]f32 @+17728 (64)
#define EDGE_SMEM (OFF_WAVE + 4 * WAVE_STRIDE)   // 196096 B (< 320KB WGP LDS)

#define OUT_SMEM  (128*128*2 + 4*16*128*2)       // 49152 B

// lrelu(x) = x>0 ? x : 0.01x  ==  max(x, 0.01x)  -> v_dual_mul + v_dual_max_num
__device__ __forceinline__ float lrelu(float x){ return __builtin_fmaxf(x, 0.01f * x); }

// A-matrix 16x32 f16 tile from LDS row-major [16][stride]:
// lanes 0-15: M=lane, halves = K {kt*32+0..7, kt*32+16..23}
// lanes16-31: M=lane-16, halves = K {kt*32+8..15, kt*32+24..31}
__device__ __forceinline__ v16h load_a_tile(const h16* base, int stride, int kt, int lane){
  const int m  = lane & 15;
  const int ko = (lane >> 4) << 3;
  const h16* p = base + m * stride + kt * 32 + ko;
  v8h lo = *(const v8h*)(p);        // 16B-aligned ds_load_b128
  v8h hi = *(const v8h*)(p + 16);
  v16h a;
#pragma unroll
  for (int i = 0; i < 8; ++i){ a[i] = lo[i]; a[i+8] = hi[i]; }
  return a;
}

// B-matrix 32x16 f16 tile from LDS-resident W^T stored [N][K]:
// lane n = nt*16 + (lane&15); lanes 0-15 hold K {kt*32+0..7, kt*32+16..23}, lanes16-31 K {kt*32+8..15, kt*32+24..31}
__device__ __forceinline__ v16h load_b_tile(const h16* wt, int K, int nt, int kt, int lane){
  const int n  = nt * 16 + (lane & 15);
  const int kb = kt * 32 + ((lane >> 4) << 3);
  const h16* p = wt + n * K + kb;
  v8h lo = *(const v8h*)(p);
  v8h hi = *(const v8h*)(p + 16);
  v16h b;
#pragma unroll
  for (int i = 0; i < 8; ++i){ b[i] = lo[i]; b[i+8] = hi[i]; }
  return b;
}

__device__ __forceinline__ v8f wmma_f16(v16h a, v16h b, v8f c){
  return __builtin_amdgcn_wmma_f32_16x16x32_f16(false, a, false, b, (short)0, c, false, false);
}

// -------------------------------------------------------------------- zero --
__global__ void zero_ws_kernel(float* p, size_t n){
  for (size_t i = (size_t)blockIdx.x * blockDim.x + threadIdx.x; i < n;
       i += (size_t)gridDim.x * blockDim.x)
    p[i] = 0.0f;
}

// -------------------------------------------------------------------- edge --
__global__ void __launch_bounds__(BLK)
edge_kernel(const float* __restrict__ hidden, const float* __restrict__ rela,
            const float* __restrict__ pe,
            const float* __restrict__ W1, const float* __restrict__ b1,
            const float* __restrict__ W2, const float* __restrict__ b2,
            const float* __restrict__ Ws, const float* __restrict__ Wr,
            const float* __restrict__ Wqr, const float* __restrict__ bqr,
            const float* __restrict__ w_alpha,
            const int* __restrict__ q_rel, const int* __restrict__ edges,
            float* __restrict__ up, float* __restrict__ bot)
{
  extern __shared__ char smem[];
  float* s_b1  = (float*)(smem + OFF_B1);
  float* s_b2  = (float*)(smem + OFF_B2);
  float* s_bqr = (float*)(smem + OFF_BQR);
  float* s_wa  = (float*)(smem + OFF_WA);
  h16* s_w1t = (h16*)(smem + OFF_W1T);
  h16* s_w2t = (h16*)(smem + OFF_W2T);
  h16* s_wst = (h16*)(smem + OFF_WST);
  h16* s_wrt = (h16*)(smem + OFF_WRT);
  h16* s_wqt = (h16*)(smem + OFF_WQT);

  const int tid  = threadIdx.x;
  const int wave = tid >> 5;
  const int lane = tid & 31;

  // cooperative weight staging, once per block: global [K][N] -> LDS f16 [N][K]
  for (int i = tid; i < 128; i += BLK){ s_b1[i] = b1[i]; s_b2[i] = b2[i]; }
  for (int i = tid; i < 64;  i += BLK){ s_bqr[i] = bqr[i]; s_wa[i] = w_alpha[i]; }
  for (int i = tid; i < 160*128; i += BLK){ int k = i >> 7, n = i & 127; s_w1t[n*160 + k] = (h16)W1[i]; }
  for (int i = tid; i < 128*128; i += BLK){ int k = i >> 7, n = i & 127; s_w2t[n*128 + k] = (h16)W2[i]; }
  for (int i = tid; i < 128*64;  i += BLK){
    int k = i >> 6, n = i & 63;
    s_wst[n*128 + k] = (h16)Ws[i];
    s_wrt[n*128 + k] = (h16)Wr[i];
    s_wqt[n*128 + k] = (h16)Wqr[i];
  }
  __syncthreads();

  char* wbase = smem + OFF_WAVE + wave * WAVE_STRIDE;
  h16*   comp    = (h16*)(wbase + 0);       // [16][160]; cols 0..127 become hr later
  h16*   hbuf    = (h16*)(wbase + 5120);    // [16][128]
  h16*   hsbuf   = (h16*)(wbase + 9216);    // [16][128]
  h16*   hqrbuf  = (h16*)(wbase + 13312);   // [16][128]
  int*   s_int   = (int*)(wbase + 17408);   // sub[16] obj[16] rel[16] tim[16] qrel[16]
  float* s_alpha = (float*)(wbase + 17728); // [16]

  const int mbase = (lane >> 4) * 8;   // C rows held by this lane
  const int ncol  = lane & 15;         // C column within tile

  for (int it = 0; it < EDGE_ITERS; ++it){
    const int tile = (blockIdx.x * EDGE_ITERS + it) * 4 + wave;  // 16-edge tile
    const int e0   = tile * 16;

    if (lane < 16){
      const int* ep = edges + (size_t)(e0 + lane) * 7;
      s_int[lane]      = ep[4];            // sub
      s_int[16 + lane] = ep[5];            // obj
      s_int[32 + lane] = ep[2];            // rel
      s_int[48 + lane] = ep[6];            // tim
      s_int[64 + lane] = q_rel[ep[0]];     // q_rel[r_idx]
      s_alpha[lane] = 0.0f;
    }
    __syncthreads();

    // gather comp = [rel_e | time_e] as f16 : 16*160 elems
#pragma unroll 4
    for (int i = 0; i < 80; ++i){
      int idx = i * 32 + lane;
      int e = idx / 160, c = idx - e * 160;
      float v = (c < 128) ? rela[(size_t)s_int[32 + e] * 128 + c]
                          : pe  [(size_t)s_int[48 + e] * 32 + (c - 128)];
      comp[e * 160 + c] = (h16)v;
    }
    // gather hs, hqr : 16*128 elems each
#pragma unroll 4
    for (int i = 0; i < 64; ++i){
      int idx = i * 32 + lane;
      int e = idx >> 7, c = idx & 127;
      hsbuf[idx]  = (h16)hidden[(size_t)s_int[e]      * 128 + c];
      hqrbuf[idx] = (h16)rela  [(size_t)s_int[64 + e] * 128 + c];
    }
    __syncthreads();

    // GEMM1: h = lrelu(comp @ W1 + b1)   [16x160]x[160x128]
    // kt-outer / nt-inner: A loaded once per k-tile, 8 live accumulators
    {
      v8f acc[8] = {};
#pragma unroll
      for (int kt = 0; kt < 5; ++kt){
        v16h a = load_a_tile(comp, 160, kt, lane);
#pragma unroll
        for (int nt = 0; nt < 8; ++nt)
          acc[nt] = wmma_f16(a, load_b_tile(s_w1t, 160, nt, kt, lane), acc[nt]);
      }
#pragma unroll
      for (int nt = 0; nt < 8; ++nt){
        int n = nt * 16 + ncol;
        float bb = s_b1[n];
#pragma unroll
        for (int r = 0; r < 8; ++r)
          hbuf[(mbase + r) * 128 + n] = (h16)lrelu(acc[nt][r] + bb);
      }
    }
    __syncthreads();

    // GEMM2: hr = lrelu(h @ W2 + b2) + rel_e ; hr overwrites comp cols 0..127
    {
      v8f acc[8] = {};
#pragma unroll
      for (int kt = 0; kt < 4; ++kt){
        v16h a = load_a_tile(hbuf, 128, kt, lane);
#pragma unroll
        for (int nt = 0; nt < 8; ++nt)
          acc[nt] = wmma_f16(a, load_b_tile(s_w2t, 128, nt, kt, lane), acc[nt]);
      }
#pragma unroll
      for (int nt = 0; nt < 8; ++nt){
        int n = nt * 16 + ncol;
        float bb = s_b2[n];
#pragma unroll
        for (int r = 0; r < 8; ++r){
          int row = mbase + r;
          float relv = (float)comp[row * 160 + n];
          comp[row * 160 + n] = (h16)(lrelu(acc[nt][r] + bb) + relv);
        }
      }
    }
    __syncthreads();

    // attention: lrelu(hs@Ws + hr@Wr + hqr@Wqr + bqr) . w_alpha  -> alpha[16]
    {
      v8f acc[4] = {};
#pragma unroll
      for (int kt = 0; kt < 4; ++kt){
        v16h a = load_a_tile(hsbuf, 128, kt, lane);
#pragma unroll
        for (int nt = 0; nt < 4; ++nt)
          acc[nt] = wmma_f16(a, load_b_tile(s_wst, 128, nt, kt, lane), acc[nt]);
      }
#pragma unroll
      for (int kt = 0; kt < 4; ++kt){
        v16h a = load_a_tile(comp, 160, kt, lane);   // hr
#pragma unroll
        for (int nt = 0; nt < 4; ++nt)
          acc[nt] = wmma_f16(a, load_b_tile(s_wrt, 128, nt, kt, lane), acc[nt]);
      }
#pragma unroll
      for (int kt = 0; kt < 4; ++kt){
        v16h a = load_a_tile(hqrbuf, 128, kt, lane);
#pragma unroll
        for (int nt = 0; nt < 4; ++nt)
          acc[nt] = wmma_f16(a, load_b_tile(s_wqt, 128, nt, kt, lane), acc[nt]);
      }
#pragma unroll
      for (int nt = 0; nt < 4; ++nt){
        int n = nt * 16 + ncol;
        float bb = s_bqr[n], wa = s_wa[n];
#pragma unroll
        for (int r = 0; r < 8; ++r)
          atomicAdd(&s_alpha[mbase + r], lrelu(acc[nt][r] + bb) * wa);
      }
    }
    __syncthreads();

    if (lane < 16){
      float ea = __expf(s_alpha[lane]);
      s_alpha[lane] = ea;
      atomicAdd(&bot[s_int[16 + lane]], ea);
    }
    __syncthreads();

    // up[obj] += ea * (hs + hr)
#pragma unroll 4
    for (int i = 0; i < 64; ++i){
      int idx = i * 32 + lane;
      int e = idx >> 7, c = idx & 127;
      float msg = (float)hsbuf[idx] + (float)comp[e * 160 + c];
      atomicAdd(&up[(size_t)s_int[16 + e] * 128 + c], s_alpha[e] * msg);
    }
    __syncthreads();   // staging buffers reused next iteration
  }
}

// --------------------------------------------------------------------- out --
__global__ void __launch_bounds__(BLK)
out_kernel(const float* __restrict__ up, const float* __restrict__ bot,
           const float* __restrict__ Wh, float* __restrict__ out)
{
  extern __shared__ char smem[];
  h16* s_wht = (h16*)smem;                       // Wh^T [128][128] f16
  const int tid  = threadIdx.x;
  const int wave = tid >> 5;
  const int lane = tid & 31;

  for (int i = tid; i < 128*128; i += BLK){ int k = i >> 7, n = i & 127; s_wht[n*128 + k] = (h16)Wh[i]; }
  __syncthreads();

  h16* abuf = (h16*)(smem + 128*128*2 + wave * (16*128*2));
  const int mbase = (lane >> 4) * 8;
  const int ncol  = lane & 15;

  // uniform trip count (4) for every wave: OUT_BOUND = OUT_BLOCKS*4*4
  for (int tile = blockIdx.x * 4 + wave; tile < OUT_BOUND; tile += OUT_BLOCKS * 4){
    const int r0 = tile * 16;

    // stage A = (up / (bot + 1e-5)) as f16, zero-padded past N_NODE
#pragma unroll 4
    for (int i = 0; i < 64; ++i){
      int idx = i * 32 + lane;
      int rr = idx >> 7, c = idx & 127;
      int row = r0 + rr;
      float v = 0.0f;
      if (row < N_NODE) v = up[(size_t)row * 128 + c] / (bot[row] + 1e-5f);
      abuf[idx] = (h16)v;
    }
    __syncthreads();

    {
      v8f acc[8] = {};
#pragma unroll
      for (int kt = 0; kt < 4; ++kt){
        v16h a = load_a_tile(abuf, 128, kt, lane);
#pragma unroll
        for (int nt = 0; nt < 8; ++nt)
          acc[nt] = wmma_f16(a, load_b_tile(s_wht, 128, nt, kt, lane), acc[nt]);
      }
#pragma unroll
      for (int nt = 0; nt < 8; ++nt){
        int n = nt * 16 + ncol;
#pragma unroll
        for (int r = 0; r < 8; ++r){
          int row = r0 + mbase + r;
          if (row < N_NODE) out[(size_t)row * 128 + n] = acc[nt][r];
        }
      }
    }
    __syncthreads();   // abuf reused next iteration
  }
}

// ------------------------------------------------------------------ launch --
extern "C" void kernel_launch(void* const* d_in, const int* in_sizes, int n_in,
                              void* d_out, int out_size, void* d_ws, size_t ws_size,
                              hipStream_t stream) {
  const float* hidden  = (const float*)d_in[0];
  const float* rela    = (const float*)d_in[1];
  const float* pe      = (const float*)d_in[2];
  const float* W1      = (const float*)d_in[3];
  const float* b1      = (const float*)d_in[4];
  const float* W2      = (const float*)d_in[5];
  const float* b2      = (const float*)d_in[6];
  const float* Ws      = (const float*)d_in[7];
  const float* Wr      = (const float*)d_in[8];
  const float* Wqr     = (const float*)d_in[9];
  const float* bqr     = (const float*)d_in[10];
  const float* w_alpha = (const float*)d_in[11];
  const float* Wh      = (const float*)d_in[12];
  const int*   q_rel   = (const int*)d_in[13];
  const int*   edges   = (const int*)d_in[14];
  (void)n_in; (void)in_sizes; (void)out_size;

  float* up  = (float*)d_ws;                       // [N_NODE][128]
  float* bot = up + (size_t)N_NODE * 128;          // [N_NODE]
  if (ws_size < (size_t)N_NODE * 129 * sizeof(float)) return;

  zero_ws_kernel<<<1024, 256, 0, stream>>>(up, (size_t)N_NODE * 129);

  edge_kernel<<<EDGE_BLOCKS, BLK, EDGE_SMEM, stream>>>(
      hidden, rela, pe, W1, b1, W2, b2, Ws, Wr, Wqr, bqr, w_alpha,
      q_rel, edges, up, bot);

  out_kernel<<<OUT_BLOCKS, BLK, OUT_SMEM, stream>>>(up, bot, Wh, (float*)d_out);
}
